// GAT_81217831568086
// MI455X (gfx1250) — compile-verified, matrix operand order checked by reading, not run on hardware
//
#include <hip/hip_runtime.h>

// ---------------- problem constants (match reference) ----------------
#define R_REL   2
#define IN_F    128
#define HID_F   64
#define OUT_F   64
#define H1      2
#define H2      1
#define NEG_SLOPE 0.2f

// ---------------- CDNA5 WMMA types ----------------
typedef __attribute__((ext_vector_type(16))) __bf16 v16bf;
typedef __attribute__((ext_vector_type(8)))  float  v8f;

union Frag16 { uint4 q[2]; v16bf v; };

// ---------------- small device helpers ----------------
__device__ __forceinline__ unsigned short f2bf(float f) {
  unsigned u = __float_as_uint(f);
  unsigned r = u + 0x7FFFu + ((u >> 16) & 1u);   // round-to-nearest-even
  return (unsigned short)(r >> 16);
}
__device__ __forceinline__ unsigned fkey(float f) {   // order-preserving map f32 -> u32
  unsigned u = __float_as_uint(f);
  return (u & 0x80000000u) ? ~u : (u | 0x80000000u);
}
__device__ __forceinline__ float funkey(unsigned k) {
  unsigned u = (k & 0x80000000u) ? (k & 0x7FFFFFFFu) : ~k;
  return __uint_as_float(u);
}
__device__ __forceinline__ float lrelu(float x) { return x >= 0.f ? x : NEG_SLOPE * x; }

// ---------------- prep kernels ----------------
__global__ void k_f32_to_bf16(const float* __restrict__ in,
                              unsigned short* __restrict__ out, size_t n) {
  size_t i = (size_t)blockIdx.x * blockDim.x + threadIdx.x;
  if (i < n) out[i] = f2bf(in[i]);
}

__global__ void k_relu_f32_to_bf16(const float* __restrict__ in,
                                   unsigned short* __restrict__ out, size_t n) {
  size_t i = (size_t)blockIdx.x * blockDim.x + threadIdx.x;
  if (i < n) { float v = in[i]; out[i] = f2bf(v > 0.f ? v : 0.f); }
}

// W: [K, NCOLS] f32  ->  Wt: [NCOLS, K] bf16
__global__ void k_transpose_w(const float* __restrict__ W,
                              unsigned short* __restrict__ Wt, int K, int NCOLS) {
  int i = blockIdx.x * blockDim.x + threadIdx.x;
  if (i >= K * NCOLS) return;
  int n = i / K, k = i % K;
  Wt[i] = f2bf(W[(size_t)k * NCOLS + n]);
}

__global__ void k_zero_u32(unsigned* __restrict__ p, size_t n) {
  size_t i = (size_t)blockIdx.x * blockDim.x + threadIdx.x;
  if (i < n) p[i] = 0u;
}

// acc[n, c] = sum_r bias[r, c]
__global__ void k_init_bias(float* __restrict__ acc, const float* __restrict__ b,
                            int N, int F) {
  size_t i = (size_t)blockIdx.x * blockDim.x + threadIdx.x;
  if (i >= (size_t)N * F) return;
  int c = (int)(i % F);
  float s = 0.f;
  for (int r = 0; r < R_REL; ++r) s += b[(size_t)r * F + c];
  acc[i] = s;
}

// ---------------- WMMA GEMM: C[M,NCOLS] = A[M,K] * Bt[NCOLS,K]^T ----------------
// Block = (NCOLS/16) waves; wave wv owns column tile wv. Each block sweeps RPB
// row strips of 16, keeping all B (weight) fragments resident in VGPRs.
template <int KDIM, int NCOLS, int RPB>
__global__ __launch_bounds__(2 * NCOLS)
void k_gemm_bf16_wmma(const unsigned short* __restrict__ A,
                      const unsigned short* __restrict__ Bt,
                      float* __restrict__ C, int M) {
  constexpr int KC = KDIM / 32;              // WMMA k-steps
  const int lane     = threadIdx.x & 31;
  const int wv       = threadIdx.x >> 5;
  const int colBase  = wv * 16;
  const int l16      = lane & 15;
  const int k0       = (lane < 16) ? 0 : 8;  // ISA 16-bit A/B fragment layout
  const int rowBlock = blockIdx.x * (16 * RPB);

  // load B fragments once (reused across all row strips of this block)
  Frag16 fb[KC];
  {
    const unsigned short* bp = Bt + (size_t)(colBase + l16) * KDIM + k0;
#pragma unroll
    for (int c = 0; c < KC; ++c) {
      fb[c].q[0] = *reinterpret_cast<const uint4*>(bp + 32 * c);
      fb[c].q[1] = *reinterpret_cast<const uint4*>(bp + 32 * c + 16);
    }
  }

  const bool full = (rowBlock + 16 * RPB) <= M;
  const int  rOff = (lane < 16) ? 0 : 8;

#pragma unroll
  for (int t = 0; t < RPB; ++t) {
    const int rowBase = rowBlock + 16 * t;
    if (rowBase >= M) break;

    int arow = rowBase + l16; if (arow >= M) arow = M - 1;   // keep EXEC all-ones
    const unsigned short* ap = A + (size_t)arow * KDIM + k0;

    v8f acc = {};
#pragma unroll
    for (int c = 0; c < KC; ++c) {
      Frag16 fa;
      fa.q[0] = *reinterpret_cast<const uint4*>(ap + 32 * c);        // K = 32c+k0..+7
      fa.q[1] = *reinterpret_cast<const uint4*>(ap + 32 * c + 16);   // K = 32c+k0+16..+23
      acc = __builtin_amdgcn_wmma_f32_16x16x32_bf16(false, fa.v, false, fb[c].v,
                                                    (short)0, acc, false, false);
    }

    // C layout: VGPR v -> row rowBase + v + (lane<16?0:8), col colBase + (lane&15)
    float* cp = C + (size_t)rowBase * NCOLS + colBase + l16;
    if (full) {
#pragma unroll
      for (int v = 0; v < 8; ++v) cp[(size_t)(rOff + v) * NCOLS] = acc[v];
    } else {
#pragma unroll
      for (int v = 0; v < 8; ++v) {
        int rr = rowBase + rOff + v;
        if (rr < M) cp[(size_t)(rOff + v) * NCOLS] = acc[v];
      }
    }
  }
}

// ---------------- attention dot products: el/er [N*H] ----------------
__global__ void k_attn_dots(const float* __restrict__ feat,
                            const float* __restrict__ al, const float* __restrict__ ar,
                            float* __restrict__ el, float* __restrict__ er,
                            int N, int H, int D) {
  int i = blockIdx.x * blockDim.x + threadIdx.x;   // node*H + h
  if (i >= N * H) return;
  int h = i % H;
  const float* f = feat + (size_t)(i / H) * H * D + (size_t)h * D;
  float sl = 0.f, sr = 0.f;
  for (int d = 0; d < D; d += 4) {
    float4 v = *reinterpret_cast<const float4*>(f + d);
    float4 a = *reinterpret_cast<const float4*>(al + h * D + d);
    float4 b = *reinterpret_cast<const float4*>(ar + h * D + d);
    sl += v.x * a.x + v.y * a.y + v.z * a.z + v.w * a.w;
    sr += v.x * b.x + v.y * b.y + v.z * b.z + v.w * b.w;
  }
  el[i] = sl; er[i] = sr;
}

// ---------------- edge softmax passes ----------------
__global__ void k_edge_max(const float* __restrict__ el, const float* __restrict__ er,
                           const int* __restrict__ src, const int* __restrict__ dst,
                           unsigned* __restrict__ mkey, int E, int H) {
  int e = blockIdx.x * blockDim.x + threadIdx.x;
  if (e >= E) return;
  int s = src[e], d = dst[e];
  for (int h = 0; h < H; ++h) {
    float v = lrelu(el[(size_t)s * H + h] + er[(size_t)d * H + h]);
    atomicMax(&mkey[(size_t)d * H + h], fkey(v));
  }
}

__global__ void k_edge_exp(const float* __restrict__ el, const float* __restrict__ er,
                           const int* __restrict__ src, const int* __restrict__ dst,
                           const unsigned* __restrict__ mkey, float* __restrict__ ssum,
                           float* __restrict__ exbuf, int E, int H) {
  int e = blockIdx.x * blockDim.x + threadIdx.x;
  if (e >= E) return;
  int s = src[e], d = dst[e];
  for (int h = 0; h < H; ++h) {
    float v  = lrelu(el[(size_t)s * H + h] + er[(size_t)d * H + h]);
    float m  = funkey(mkey[(size_t)d * H + h]);
    float ex = __expf(v - m);
    exbuf[(size_t)e * H + h] = ex;
    atomicAdd(&ssum[(size_t)d * H + h], ex);
  }
}

// one wave per edge: lanes stripe the H*D feature vector with vector loads
template <int H, int D>
__global__ void k_edge_scatter(const float* __restrict__ exbuf,
                               const float* __restrict__ ssum,
                               const float* __restrict__ feat,
                               const int* __restrict__ src, const int* __restrict__ dst,
                               float* __restrict__ acc, int E) {
  constexpr int F   = H * D;
  constexpr int PER = F / 32;   // floats per lane: 4 (F=128) or 2 (F=64)
  int wid  = (int)(((size_t)blockIdx.x * blockDim.x + threadIdx.x) >> 5);
  int lane = threadIdx.x & 31;
  if (wid >= E) return;
  int s = src[wid], d = dst[wid];

  const int base = lane * PER;          // PER divides D -> whole lane in one head
  const int h    = base / D;
  float sum   = ssum[(size_t)d * H + h];
  float alpha = exbuf[(size_t)wid * H + h] / (sum > 0.f ? sum : 1.f);

  const float* fp = feat + (size_t)s * F + base;
  float*       op = acc  + (size_t)d * F + base;
  if constexpr (PER == 4) {
    float4 v = *reinterpret_cast<const float4*>(fp);
    atomicAdd(op + 0, alpha * v.x);
    atomicAdd(op + 1, alpha * v.y);
    atomicAdd(op + 2, alpha * v.z);
    atomicAdd(op + 3, alpha * v.w);
  } else {
    float2 v = *reinterpret_cast<const float2*>(fp);
    atomicAdd(op + 0, alpha * v.x);
    atomicAdd(op + 1, alpha * v.y);
  }
}

// ---------------- host-side orchestration ----------------
extern "C" void kernel_launch(void* const* d_in, const int* in_sizes, int n_in,
                              void* d_out, int out_size, void* d_ws, size_t ws_size,
                              hipStream_t stream) {
  const float* x    = (const float*)d_in[0];
  const int*   esrc = (const int*)d_in[1];     // [2, R, E]
  const int*   edst = (const int*)d_in[2];
  const float* W1   = (const float*)d_in[3];   // [R, 128, 128]
  const float* al1  = (const float*)d_in[4];   // [R, 2, 64]
  const float* ar1  = (const float*)d_in[5];
  const float* b1   = (const float*)d_in[6];   // [R, 128]
  const float* W2   = (const float*)d_in[7];   // [R, 128, 64]
  const float* al2  = (const float*)d_in[8];   // [R, 1, 64]
  const float* ar2  = (const float*)d_in[9];
  const float* b2   = (const float*)d_in[10];  // [R, 64]
  float* out = (float*)d_out;

  const int N = in_sizes[0] / IN_F;
  const int E = in_sizes[1] / (2 * R_REL);

  // workspace carve-up (256B aligned)
  size_t off = 0;
  auto carve = [&](size_t bytes) -> void* {
    void* p = (char*)d_ws + off;
    off += (bytes + 255) & ~(size_t)255;
    return p;
  };
  unsigned short* xbf  = (unsigned short*)carve((size_t)N * IN_F * 2);
  unsigned short* hbf  = (unsigned short*)carve((size_t)N * (H1 * HID_F) * 2);
  unsigned short* wt1  = (unsigned short*)carve((size_t)R_REL * IN_F * (H1 * HID_F) * 2);
  unsigned short* wt2  = (unsigned short*)carve((size_t)R_REL * (H1 * HID_F) * OUT_F * 2);
  float*    feat  = (float*)carve((size_t)N * (H1 * HID_F) * 4);
  float*    el    = (float*)carve((size_t)N * H1 * 4);
  float*    er    = (float*)carve((size_t)N * H1 * 4);
  unsigned* mkey  = (unsigned*)carve((size_t)N * H1 * 4);
  float*    ssum  = (float*)carve((size_t)N * H1 * 4);
  float*    exbuf = (float*)carve((size_t)E * H1 * 4);
  float*    hacc  = (float*)carve((size_t)N * (H1 * HID_F) * 4);

  const int T = 256;
  auto blk1 = [&](size_t n) { return (unsigned)((n + T - 1) / T); };

  // ---- prep: bf16 x, transposed bf16 weights ----
  k_f32_to_bf16<<<blk1((size_t)N * IN_F), T, 0, stream>>>(x, xbf, (size_t)N * IN_F);
  for (int r = 0; r < R_REL; ++r) {
    k_transpose_w<<<blk1((size_t)IN_F * (H1 * HID_F)), T, 0, stream>>>(
        W1 + (size_t)r * IN_F * (H1 * HID_F), wt1 + (size_t)r * IN_F * (H1 * HID_F),
        IN_F, H1 * HID_F);
    k_transpose_w<<<blk1((size_t)(H1 * HID_F) * OUT_F), T, 0, stream>>>(
        W2 + (size_t)r * (H1 * HID_F) * OUT_F, wt2 + (size_t)r * (H1 * HID_F) * OUT_F,
        H1 * HID_F, OUT_F);
  }

  constexpr int RPB = 4;                       // 64 rows per block
  const unsigned gemmGrid = (unsigned)((N + 16 * RPB - 1) / (16 * RPB));

  // ================= layer 1 (H=2, D=64, F=128) =================
  k_init_bias<<<blk1((size_t)N * (H1 * HID_F)), T, 0, stream>>>(hacc, b1, N, H1 * HID_F);
  for (int r = 0; r < R_REL; ++r) {
    const int* s1 = esrc + (size_t)(0 * R_REL + r) * E;
    const int* d1 = edst + (size_t)(0 * R_REL + r) * E;

    k_gemm_bf16_wmma<IN_F, H1 * HID_F, RPB><<<gemmGrid, 2 * (H1 * HID_F), 0, stream>>>(
        xbf, wt1 + (size_t)r * IN_F * (H1 * HID_F), feat, N);

    k_attn_dots<<<blk1((size_t)N * H1), T, 0, stream>>>(
        feat, al1 + (size_t)r * H1 * HID_F, ar1 + (size_t)r * H1 * HID_F,
        el, er, N, H1, HID_F);

    k_zero_u32<<<blk1((size_t)N * H1), T, 0, stream>>>(mkey, (size_t)N * H1);
    k_zero_u32<<<blk1((size_t)N * H1), T, 0, stream>>>((unsigned*)ssum, (size_t)N * H1);

    k_edge_max<<<blk1((size_t)E), T, 0, stream>>>(el, er, s1, d1, mkey, E, H1);
    k_edge_exp<<<blk1((size_t)E), T, 0, stream>>>(el, er, s1, d1, mkey, ssum, exbuf, E, H1);
    k_edge_scatter<H1, HID_F><<<blk1((size_t)E * 32), T, 0, stream>>>(
        exbuf, ssum, feat, s1, d1, hacc, E);
  }

  // relu + convert to bf16 for layer-2 GEMM input
  k_relu_f32_to_bf16<<<blk1((size_t)N * (H1 * HID_F)), T, 0, stream>>>(
      hacc, hbf, (size_t)N * (H1 * HID_F));

  // ================= layer 2 (H=1, D=64, F=64) -> d_out =================
  k_init_bias<<<blk1((size_t)N * OUT_F), T, 0, stream>>>(out, b2, N, OUT_F);
  for (int r = 0; r < R_REL; ++r) {
    const int* s2 = esrc + (size_t)(1 * R_REL + r) * E;
    const int* d2 = edst + (size_t)(1 * R_REL + r) * E;

    k_gemm_bf16_wmma<H1 * HID_F, OUT_F, RPB><<<gemmGrid, 2 * OUT_F, 0, stream>>>(
        hbf, wt2 + (size_t)r * (H1 * HID_F) * OUT_F, feat, N);

    k_attn_dots<<<blk1((size_t)N * H2), T, 0, stream>>>(
        feat, al2 + (size_t)r * H2 * OUT_F, ar2 + (size_t)r * H2 * OUT_F,
        el, er, N, H2, OUT_F);

    k_zero_u32<<<blk1((size_t)N * H2), T, 0, stream>>>(mkey, (size_t)N * H2);
    k_zero_u32<<<blk1((size_t)N * H2), T, 0, stream>>>((unsigned*)ssum, (size_t)N * H2);

    k_edge_max<<<blk1((size_t)E), T, 0, stream>>>(el, er, s2, d2, mkey, E, H2);
    k_edge_exp<<<blk1((size_t)E), T, 0, stream>>>(el, er, s2, d2, mkey, ssum, exbuf, E, H2);
    k_edge_scatter<H2, OUT_F><<<blk1((size_t)E * 32), T, 0, stream>>>(
        exbuf, ssum, feat, s2, d2, out, E);
  }
}
